// QRNN_2044404433191
// MI455X (gfx1250) — matrix operand order repaired and checked by reading, not compiled, not run on hardware
//
#include <hip/hip_runtime.h>
#include <hip/hip_bf16.h>
#include <math.h>

// QRNN fo-pooling for MI455X (gfx1250, wave32, WMMA).
// B=8, T=4096, C=512, H=512, K=2.

#define B_   8
#define T_   4096
#define C_   512
#define H_   512
#define KW   2
#define M_   (B_ * T_)      // 32768 GEMM rows
#define KK_  (KW * C_)      // 1024 GEMM K
#define NG   3              // gates z, f, o
#define TC   128            // scan chunk length
#define NCH  (T_ / TC)      // 32 chunks per sequence

#define BROW 40             // padded LDS row stride (elements) for B tile
#define NKIT (KK_ / 32)     // 32 k-steps

#define AS1 __attribute__((address_space(1)))
#define AS3 __attribute__((address_space(3)))

#if defined(__has_builtin)
#if __has_builtin(__builtin_amdgcn_global_load_async_to_lds_b128)
#define QRNN_ASYNC_LDS 1
#endif
#endif

typedef __attribute__((ext_vector_type(16))) __bf16 v16bf;
typedef __attribute__((ext_vector_type(8)))  float  v8f;
typedef __attribute__((ext_vector_type(4)))  int    v4i;

__device__ __forceinline__ unsigned short f2bf(float f) {
    unsigned int u = __float_as_uint(f);
    unsigned int r = u + 0x7FFFu + ((u >> 16) & 1u);   // round to nearest even
    return (unsigned short)(r >> 16);
}

// ---------------------------------------------------------------------------
// Kernel 1: causal-padded bf16 copy of x.
// xb[b][t][c], t in [0, T]; t==0 row is zero, t>=1 holds x[b][t-1][c].
// Rows t and t+1 are adjacent -> width-2 conv row == contiguous K=1024 span.
// ---------------------------------------------------------------------------
__global__ void qrnn_pad_x(const float* __restrict__ x, unsigned short* __restrict__ xb,
                           size_t total) {
    size_t idx = (size_t)blockIdx.x * blockDim.x + threadIdx.x;
    if (idx >= total) return;
    int c = (int)(idx % C_);
    size_t rt = idx / C_;
    int t = (int)(rt % (T_ + 1));
    int b = (int)(rt / (T_ + 1));
    float v = (t == 0) ? 0.0f : x[((size_t)b * T_ + (t - 1)) * C_ + c];
    xb[idx] = f2bf(v);
}

// ---------------------------------------------------------------------------
// Kernel 2: repack weights f32 [K][C][H] -> bf16 WT[g][h][k*C + c]
// (K-contiguous per output column, so B-fragment dword loads are contiguous).
// ---------------------------------------------------------------------------
__global__ void qrnn_prep_w(const float* __restrict__ zk, const float* __restrict__ fk,
                            const float* __restrict__ ok, unsigned short* __restrict__ wt,
                            size_t total) {
    size_t idx = (size_t)blockIdx.x * blockDim.x + threadIdx.x;
    if (idx >= total) return;
    int k2 = (int)(idx % KK_);
    int h  = (int)((idx / KK_) % H_);
    int g  = (int)(idx / ((size_t)KK_ * H_));
    const float* src = (g == 0) ? zk : ((g == 1) ? fk : ok);
    // kernel flat layout: (k*C + c)*H + h == k2*H + h
    wt[idx] = f2bf(src[(size_t)k2 * H_ + h]);
}

// ---------------------------------------------------------------------------
// Stage one 16-byte chunk of the 64x32 B k-slice into LDS.
// Prefer the gfx1250 async Global->LDS path (ASYNCcnt, no VGPR round trip);
// fall back to load+ds_store if the builtin is unavailable.
// ---------------------------------------------------------------------------
__device__ __forceinline__ void stage_B(const unsigned short* __restrict__ wt,
                                        unsigned short* dst, int n0, int sn, int sk, int kk) {
    const unsigned short* gp = wt + (size_t)(n0 + sn) * KK_ + kk + sk;
    unsigned short* lp = dst + sn * BROW + sk;
#ifdef QRNN_ASYNC_LDS
    __builtin_amdgcn_global_load_async_to_lds_b128(
        (AS1 v4i*)gp, (AS3 v4i*)lp, 0, 0);
#else
    *(uint4*)lp = *(const uint4*)gp;
#endif
}

// ---------------------------------------------------------------------------
// Kernel 3: fused GEMM + bias + activation via v_wmma_f32_16x16x32_bf16.
// 256-thread block = 8 waves arranged 4(M) x 2(N); macro tile 256M x 64N.
// The 64x32 B k-slice is double-buffered in LDS (stage slice i+1 async while
// computing slice i; one barrier per k-step). Padded LDS rows keep the
// per-lane 2x ds_load_b128 fragment reads bank-conflict free. A fragments
// stream straight from the L2-resident bf16 x copy via global_load_b128.
// Writes activated z, f, o into gates[3][M][H].
// ---------------------------------------------------------------------------
__global__ void __launch_bounds__(256)
qrnn_gemm(const unsigned short* __restrict__ xb, const unsigned short* __restrict__ wt,
          const float* __restrict__ zb, const float* __restrict__ fb,
          const float* __restrict__ ob, float* __restrict__ gates) {
    __shared__ __align__(16) unsigned short Bs[2][64 * BROW];   // 2 x 5 KB

    const int tid  = threadIdx.x;
    const int lane = tid & 31;
    const int wv   = tid >> 5;          // wave id 0..7
    const int lm   = lane & 15;         // m (A) / n (B,C,D) within 16x16 tile
    const int hi   = lane >> 4;         // K-half selector

    const int n0 = blockIdx.x * 64;     // macro N origin
    const int m0 = blockIdx.y * 256;    // macro M origin
    const int nw = (wv & 1) * 32;       // wave N offset within macro
    const int mw = (wv >> 1) * 64;      // wave M offset within macro
    const int gate = (n0 + nw) >> 9;    // 32-wide N tile never crosses a gate

    // B staging assignment: 64 rows x 32 elems = 8 bf16 (16B) per thread.
    const int sn = tid >> 2;            // staged row 0..63
    const int sk = (tid & 3) * 8;       // staged k offset 0..24

    union ABu { v16bf v; unsigned int u[8]; };

    v8f acc[4][2] = {};

    // Per-lane A row base (element offset into xb). 256-row macro tiles never
    // cross a batch boundary (4096 % 256 == 0).
    size_t arow[4];
#pragma unroll
    for (int i = 0; i < 4; ++i) {
        int r = m0 + mw + 16 * i + lm;
        int b = r >> 12;                // r / T_
        int t = r & (T_ - 1);
        arow[i] = ((size_t)b * (T_ + 1) + t) * C_;
    }

    // Prologue: stage k-slice 0 into buffer 0.
    stage_B(wt, &Bs[0][0], n0, sn, sk, 0);

    for (int it = 0; it < NKIT; ++it) {
        const int kk = it * 32;
        const unsigned short* bufc = &Bs[it & 1][0];

#ifdef QRNN_ASYNC_LDS
        // My async stage of the current buffer has landed in LDS.
        asm volatile("s_wait_asynccnt 0x0" ::: "memory");
#endif
        // After this barrier: (a) current buffer fully staged by all threads,
        // (b) all threads finished reading the other buffer last iteration,
        // so it is safe to overwrite it now.
        __syncthreads();

        if (it + 1 < NKIT)
            stage_B(wt, &Bs[(it + 1) & 1][0], n0, sn, sk, kk + 32);

        // ---- load fragments ----
        ABu a[4], bfr[2];
        // A 16x32 bf16 layout: VGPR j<4: K = kk + 2j + 8*hi (+pos);
        //                      VGPR j>=4: K = kk + 16 + 2(j-4) + 8*hi (+pos).
#pragma unroll
        for (int j = 0; j < 8; ++j) {
            int ka = kk + ((j & 4) ? 16 : 0) + ((j & 3) * 2) + hi * 8;
#pragma unroll
            for (int i = 0; i < 4; ++i)
                a[i].u[j] = *(const unsigned int*)(xb + arow[i] + ka);
        }
        // B 32x16 bf16 layout: same K order as A (VGPR j<4: K=2j+8*hi; j>=4: K=16+2(j-4)+8*hi).
#pragma unroll
        for (int j2 = 0; j2 < 2; ++j2) {
            int nrow = nw + 16 * j2 + lm;
#pragma unroll
            for (int j = 0; j < 8; ++j)
                bfr[j2].u[j] = *(const unsigned int*)(bufc + nrow * BROW + ((j & 4) ? 16 : 0) + hi * 8 + 2 * (j & 3));
        }

#pragma unroll
        for (int i = 0; i < 4; ++i)
#pragma unroll
            for (int j = 0; j < 2; ++j)
                acc[i][j] = __builtin_amdgcn_wmma_f32_16x16x32_bf16(
                    false, a[i].v, false, bfr[j].v,
                    (short)0, acc[i][j], false, false);
    }

    const float* bias = (gate == 0) ? zb : ((gate == 1) ? fb : ob);
#pragma unroll
    for (int j = 0; j < 2; ++j) {
        int n  = n0 + nw + 16 * j + lm;
        int nl = n - gate * 512;
        float bv = bias[nl];
#pragma unroll
        for (int i = 0; i < 4; ++i) {
#pragma unroll
            for (int r = 0; r < 8; ++r) {
                int row = m0 + mw + 16 * i + r + 8 * hi;   // C/D f32 layout
                float v = acc[i][j][r] + bv;
                v = (gate == 0) ? tanhf(v) : 1.0f / (1.0f + __expf(-v));
                gates[((size_t)gate * M_ + row) * H_ + nl] = v;
            }
        }
    }
}

// ---------------------------------------------------------------------------
// Kernel 4: per-chunk scan. For each (b, chunk, h): A = prod(f), Bv = h value
// after the chunk starting from h=0. 131072 parallel tasks of 128 steps.
// ---------------------------------------------------------------------------
__global__ void qrnn_chunk_scan(const float* __restrict__ gates,
                                float* __restrict__ cA, float* __restrict__ cB) {
    int tid = blockIdx.x * blockDim.x + threadIdx.x;
    int h = tid & (H_ - 1);
    int rest = tid >> 9;
    int chunk = rest & (NCH - 1);
    int b = rest >> 5;
    const float* Z = gates;
    const float* F = gates + (size_t)M_ * H_;
    size_t base = ((size_t)b * T_ + (size_t)chunk * TC) * H_ + h;
    float a = 1.0f, hacc = 0.0f;
    for (int i = 0; i < TC; ++i) {
        float f = F[base + (size_t)i * H_];
        float z = Z[base + (size_t)i * H_];
        hacc = f * hacc + (1.0f - f) * z;    // zin = (1-f)*z
        a *= f;
    }
    int ci = (b * NCH + chunk) * H_ + h;
    cA[ci] = a;
    cB[ci] = hacc;
}

// ---------------------------------------------------------------------------
// Kernel 5: per-sequence prefix over the 32 chunk states -> chunk-entry h0.
// ---------------------------------------------------------------------------
__global__ void qrnn_chunk_prefix(const float* __restrict__ cA, const float* __restrict__ cB,
                                  float* __restrict__ h0) {
    int tid = blockIdx.x * blockDim.x + threadIdx.x;   // B*H threads
    int h = tid & (H_ - 1);
    int b = tid >> 9;
    float hp = 0.0f;
    for (int chunk = 0; chunk < NCH; ++chunk) {
        int ci = (b * NCH + chunk) * H_ + h;
        h0[ci] = hp;
        hp = cA[ci] * hp + cB[ci];
    }
}

// ---------------------------------------------------------------------------
// Kernel 6: final pass — recompute chunk with correct h0, emit out = h * o.
// ---------------------------------------------------------------------------
__global__ void qrnn_final(const float* __restrict__ gates, const float* __restrict__ h0,
                           float* __restrict__ out) {
    int tid = blockIdx.x * blockDim.x + threadIdx.x;
    int h = tid & (H_ - 1);
    int rest = tid >> 9;
    int chunk = rest & (NCH - 1);
    int b = rest >> 5;
    const float* Z = gates;
    const float* F = gates + (size_t)M_ * H_;
    const float* O = gates + 2 * (size_t)M_ * H_;
    size_t base = ((size_t)b * T_ + (size_t)chunk * TC) * H_ + h;
    float hv = h0[(b * NCH + chunk) * H_ + h];
    for (int i = 0; i < TC; ++i) {
        size_t p = base + (size_t)i * H_;
        float f = F[p];
        float z = Z[p];
        hv = f * hv + (1.0f - f) * z;
        out[p] = hv * O[p];
    }
}

// ---------------------------------------------------------------------------
extern "C" void kernel_launch(void* const* d_in, const int* in_sizes, int n_in,
                              void* d_out, int out_size, void* d_ws, size_t ws_size,
                              hipStream_t stream) {
    (void)in_sizes; (void)n_in; (void)out_size; (void)ws_size;
    const float* x  = (const float*)d_in[0];
    const float* zk = (const float*)d_in[1];
    const float* zbp= (const float*)d_in[2];
    const float* fk = (const float*)d_in[3];
    const float* fbp= (const float*)d_in[4];
    const float* ok = (const float*)d_in[5];
    const float* obp= (const float*)d_in[6];
    float* out = (float*)d_out;

    char* ws = (char*)d_ws;
    size_t off = 0;
    auto alloc = [&](size_t bytes) {
        char* p = ws + off;
        off = (off + bytes + 255) & ~(size_t)255;
        return p;
    };
    unsigned short* xb    = (unsigned short*)alloc((size_t)B_ * (T_ + 1) * C_ * 2); // 33.6 MB
    unsigned short* wt    = (unsigned short*)alloc((size_t)NG * H_ * KK_ * 2);      //  3.1 MB
    float*          gates = (float*)alloc((size_t)NG * M_ * H_ * 4);                // 201 MB
    float*          cA    = (float*)alloc((size_t)B_ * NCH * H_ * 4);
    float*          cB    = (float*)alloc((size_t)B_ * NCH * H_ * 4);
    float*          h0    = (float*)alloc((size_t)B_ * NCH * H_ * 4);

    // 1) causal-padded bf16 x
    {
        size_t total = (size_t)B_ * (T_ + 1) * C_;
        int blocks = (int)((total + 255) / 256);
        qrnn_pad_x<<<blocks, 256, 0, stream>>>(x, xb, total);
    }
    // 2) bf16 transposed weights
    {
        size_t total = (size_t)NG * H_ * KK_;
        int blocks = (int)((total + 255) / 256);
        qrnn_prep_w<<<blocks, 256, 0, stream>>>(zk, fk, ok, wt, total);
    }
    // 3) WMMA GEMM + activations: grid (N macro = 1536/64, M macro = 32768/256)
    {
        dim3 grid(NG * H_ / 64, M_ / 256);
        qrnn_gemm<<<grid, 256, 0, stream>>>(xb, wt, zbp, fbp, obp, gates);
    }
    // 4) per-chunk scan
    {
        int total = B_ * NCH * H_;        // 131072
        qrnn_chunk_scan<<<total / 256, 256, 0, stream>>>(gates, cA, cB);
    }
    // 5) per-sequence chunk prefix
    {
        int total = B_ * H_;              // 4096
        qrnn_chunk_prefix<<<total / 256, 256, 0, stream>>>(cA, cB, h0);
    }
    // 6) final recompute + output gating
    {
        int total = B_ * NCH * H_;
        qrnn_final<<<total / 256, 256, 0, stream>>>(gates, h0, out);
    }
}
